// LateralCoupling_2705829397193
// MI455X (gfx1250) — matrix operand order repaired and checked by reading, not compiled, not run on hardware
//
#include <hip/hip_runtime.h>
#include <hip/hip_bf16.h>

typedef __bf16 v16bf  __attribute__((ext_vector_type(16)));
typedef __bf16 bf16x8 __attribute__((ext_vector_type(8)));
typedef __bf16 bf16x4 __attribute__((ext_vector_type(4)));
typedef float  v8f    __attribute__((ext_vector_type(8)));

#define BT   (8 * 2048)   // B*T tokens
#define PP   13           // P
#define DD   256          // D
#define TOK  4            // tokens per block
#define NW   4            // waves per block
#define SQ   264          // bf16 row stride for hb / q / k(ctx)   (16B-aligned rows)
#define SVT  16           // bf16 row stride for v^T rows (one e, 16 q' slots)

// LDS element counts
#define HB_ELT   (TOK * 16 * SQ)
#define VT_ELT   (TOK * 256 * SVT)
#define AT_ELT   (TOK * 256)
#define SMEM_BYTES ((3 * HB_ELT + VT_ELT + AT_ELT) * 2 + 256 * 4)

// ---------------- weight prepack: fp32 (E,D) -> bf16 WMMA B-fragments ----------------
// B-frag (K=32,N=16): lane = n + 16*half; element i = B[k = 16*half + i, n],  B[d,e] = W[e,d]
// dst index: (((proj*16 + nt)*8 + ks)*32 + lane)*16 + i
__global__ void prepack_weights(const float* __restrict__ Wq, const float* __restrict__ Wk,
                                const float* __restrict__ Wv, const float* __restrict__ Wo,
                                __bf16* __restrict__ out) {
    int tid  = blockIdx.x * blockDim.x + threadIdx.x;   // 0..16383
    int lane = tid & 31;
    int ks   = (tid >> 5) & 7;
    int nt   = (tid >> 8) & 15;
    int proj = (tid >> 12) & 3;
    const float* W = (proj == 0) ? Wq : (proj == 1) ? Wk : (proj == 2) ? Wv : Wo;
    int n    = nt * 16 + (lane & 15);
    int half = lane >> 4;
    __bf16* dst = out + (size_t)tid * 16;
#pragma unroll
    for (int i = 0; i < 16; ++i)
        dst[i] = (__bf16)W[n * DD + ks * 32 + half * 16 + i];
}

// A-fragment (16x32 bf16) from a row-major bf16 row: two aligned 16B runs
__device__ __forceinline__ v16bf load_afrag(const __bf16* row, int ks, int half) {
    bf16x8 lo = *(const bf16x8*)(row + ks * 32 + 16 * half);
    bf16x8 hi = *(const bf16x8*)(row + ks * 32 + 16 * half + 8);
    return __builtin_shufflevector(lo, hi, 0,1,2,3,4,5,6,7,8,9,10,11,12,13,14,15);
}
// 16 contiguous bf16 (two aligned 16B halves)
__device__ __forceinline__ v16bf load_b16(const __bf16* p) {
    bf16x8 lo = *(const bf16x8*)(p);
    bf16x8 hi = *(const bf16x8*)(p + 8);
    return __builtin_shufflevector(lo, hi, 0,1,2,3,4,5,6,7,8,9,10,11,12,13,14,15);
}

// ---------------- main kernel: 4 waves, 4 tokens per block ----------------
__global__ void __launch_bounds__(NW * 32)
lateral_coupling_kernel(const float* __restrict__ h,
                        const __bf16* __restrict__ wfrags,
                        const float* __restrict__ Wlat,
                        const float* __restrict__ gate,
                        float* __restrict__ out) {
    extern __shared__ char smem[];
    __bf16* sh_hb   = (__bf16*)smem;              // [TOK][16][SQ] bf16 h tiles
    __bf16* sh_q    = sh_hb + HB_ELT;             // [TOK][16][SQ] q
    __bf16* sh_k    = sh_q  + HB_ELT;             // [TOK][16][SQ] k, reused for ctx
    __bf16* sh_vt   = sh_k  + HB_ELT;             // [TOK][256][SVT] v^T: [e][q']
    __bf16* sh_attn = sh_vt + VT_ELT;             // [TOK][16][16]
    float*  sh_wlat = (float*)(sh_attn + AT_ELT); // 16x16 padded W_lat

    const int lane = threadIdx.x & 31;
    const int w    = threadIdx.x >> 5;            // wave id 0..3
    const int m    = lane & 15;
    const int half = lane >> 4;

    __builtin_prefetch(wfrags, 0, 3);

    // W_lat -> LDS, padded 16x16
    for (int i = threadIdx.x; i < 256; i += NW * 32) {
        int p = i >> 4, q = i & 15;
        sh_wlat[i] = (p < PP && q < PP) ? Wlat[p * PP + q] : 0.0f;
    }

    // ---- P0: wave w stages token w (fp32 -> bf16 row-major, rows 13..15 zero) ----
    {
        const float* hp = h + (size_t)(blockIdx.x * TOK + w) * (PP * DD);
        __bf16* hbw = sh_hb + w * 16 * SQ;
        bf16x4 z = {};
#pragma unroll 1
        for (int p = 0; p < 16; ++p) {
            bf16x4 c0 = z, c1 = z;
            if (p < PP) {
                const float4* src = (const float4*)(hp + p * DD);
                float4 v0 = src[lane], v1 = src[lane + 32];
                c0 = (bf16x4){(__bf16)v0.x, (__bf16)v0.y, (__bf16)v0.z, (__bf16)v0.w};
                c1 = (bf16x4){(__bf16)v1.x, (__bf16)v1.y, (__bf16)v1.z, (__bf16)v1.w};
            }
            *(bf16x4*)(hbw + p * SQ + 4 * lane)       = c0;
            *(bf16x4*)(hbw + p * SQ + 128 + 4 * lane) = c1;
        }
    }
    __syncthreads();

    // ---- P1: q/k/v projections; wave w owns nt = 4w..4w+3, loops all 4 tokens ----
    // proj loop fully unrolled so each projection's store path is specialized
    // (no per-iteration branches between WMMA chains)
#pragma unroll
    for (int proj = 0; proj < 3; ++proj) {
#pragma unroll 1
        for (int ntl = 0; ntl < 4; ++ntl) {
            const int nt = w * 4 + ntl;
            v16bf bw[8];
#pragma unroll
            for (int ks = 0; ks < 8; ++ks)
                bw[ks] = *(const v16bf*)(wfrags +
                    ((((size_t)proj * 16 + nt) * 8 + ks) * 32 + lane) * 16);
            const int col = nt * 16 + m;
#pragma unroll 1
            for (int tok = 0; tok < TOK; ++tok) {
                const __bf16* arow = sh_hb + tok * 16 * SQ + m * SQ;
                v8f acc = {};
#pragma unroll
                for (int ks = 0; ks < 8; ++ks)
                    acc = __builtin_amdgcn_wmma_f32_16x16x32_bf16(
                        false, load_afrag(arow, ks, half), false, bw[ks],
                        (short)0, acc, false, false);
                if (proj == 0) {
                    __bf16* d = sh_q + tok * 16 * SQ;
#pragma unroll
                    for (int r = 0; r < 8; ++r) d[(r + 8 * half) * SQ + col] = (__bf16)acc[r];
                } else if (proj == 1) {
                    __bf16* d = sh_k + tok * 16 * SQ;
#pragma unroll
                    for (int r = 0; r < 8; ++r) d[(r + 8 * half) * SQ + col] = (__bf16)acc[r];
                } else {
                    __bf16* d = sh_vt + tok * 256 * SVT;
#pragma unroll
                    for (int r = 0; r < 8; ++r) d[col * SVT + (r + 8 * half)] = (__bf16)acc[r];
                }
            }
        }
    }
    __syncthreads();

    // ---- P2: attention; wave w handles token w entirely ----
    {
        const int tok = w;
        const __bf16* qrow = sh_q + tok * 16 * SQ + m * SQ;
        const __bf16* krow = sh_k + tok * 16 * SQ + m * SQ;
        v8f s = {};
#pragma unroll
        for (int ks = 0; ks < 8; ++ks)
            s = __builtin_amdgcn_wmma_f32_16x16x32_bf16(
                false, load_afrag(qrow, ks, half), false,
                load_b16(krow + ks * 32 + 16 * half), (short)0, s, false, false);

        // softmax over columns (q' axis), 13 valid of 16
        const float scale = 0.0625f;       // 1/sqrt(256)
        const bool valid = (m < PP);
        __bf16* at = sh_attn + tok * 256;
#pragma unroll
        for (int r = 0; r < 8; ++r) {
            float v = valid ? s[r] * scale : -3.4e38f;
            float mv = v;
#pragma unroll
            for (int off = 1; off < 16; off <<= 1) mv = fmaxf(mv, __shfl_xor(mv, off, 32));
            float e = valid ? __expf(v - mv) : 0.0f;
            float sm = e;
#pragma unroll
            for (int off = 1; off < 16; off <<= 1) sm += __shfl_xor(sm, off, 32);
            at[(r + 8 * half) * 16 + m] = (__bf16)(e / sm);
        }

        // attn as A-fragment (K padded 13->32 with zeros)
        v16bf aat = {};
#pragma unroll
        for (int i = 0; i < 16; ++i) {
            int K = half * 16 + i;
            aat[i] = (K < PP) ? at[m * 16 + K] : (__bf16)0.0f;
        }

        // ctx = attn @ v ; write ctx into sh_k[tok] (k is dead)
        __bf16* ctxb = sh_k + tok * 16 * SQ;
        const __bf16* vtb = sh_vt + tok * 256 * SVT;
#pragma unroll 1
        for (int nt = 0; nt < 16; ++nt) {
            v16bf b = {};
            if (half == 0) b = load_b16(vtb + (nt * 16 + m) * SVT);
            v8f cz = {};
            v8f c = __builtin_amdgcn_wmma_f32_16x16x32_bf16(
                false, aat, false, b, (short)0, cz, false, false);
            const int col = nt * 16 + m;
#pragma unroll
            for (int r = 0; r < 8; ++r) ctxb[(r + 8 * half) * SQ + col] = (__bf16)c[r];
        }
    }
    __syncthreads();

    // ---- P4: out = residual + sigmoid(gate) * (ctx @ Wo^T) ----
    const float g = 1.0f / (1.0f + __expf(-gate[0]));
#pragma unroll 1
    for (int ntl = 0; ntl < 4; ++ntl) {
        const int nt = w * 4 + ntl;
        v16bf bw[8];
#pragma unroll
        for (int ks = 0; ks < 8; ++ks)
            bw[ks] = *(const v16bf*)(wfrags +
                (((size_t)(3 * 16 + nt) * 8 + ks) * 32 + lane) * 16);
        const int e = nt * 16 + m;
#pragma unroll 1
        for (int tok = 0; tok < TOK; ++tok) {
            const __bf16* crow = sh_k + tok * 16 * SQ + m * SQ;   // ctx
            v8f acc = {};
#pragma unroll
            for (int ks = 0; ks < 8; ++ks)
                acc = __builtin_amdgcn_wmma_f32_16x16x32_bf16(
                    false, load_afrag(crow, ks, half), false, bw[ks],
                    (short)0, acc, false, false);
            // lateral residual from staged h
            const __bf16* hbt = sh_hb + tok * 16 * SQ;
            float res[8];
#pragma unroll
            for (int r = 0; r < 8; ++r) res[r] = 0.0f;
#pragma unroll
            for (int p = 0; p < PP; ++p) {
                float hv = (float)hbt[p * SQ + e];
#pragma unroll
                for (int r = 0; r < 8; ++r) res[r] += hv * sh_wlat[p * 16 + (r + 8 * half)];
            }
            float* op = out + (size_t)(blockIdx.x * TOK + tok) * (PP * DD);
#pragma unroll
            for (int r = 0; r < 8; ++r) {
                int qrow = r + 8 * half;
                if (qrow < PP) op[qrow * DD + e] = res[r] + g * acc[r];
            }
        }
    }
}

extern "C" void kernel_launch(void* const* d_in, const int* in_sizes, int n_in,
                              void* d_out, int out_size, void* d_ws, size_t ws_size,
                              hipStream_t stream) {
    (void)in_sizes; (void)n_in; (void)out_size; (void)ws_size;
    const float* h    = (const float*)d_in[0];
    const float* Wq   = (const float*)d_in[1];
    const float* Wk   = (const float*)d_in[2];
    const float* Wv   = (const float*)d_in[3];
    const float* Wo   = (const float*)d_in[4];
    const float* Wlat = (const float*)d_in[5];
    const float* gate = (const float*)d_in[6];
    __bf16* wfrags = (__bf16*)d_ws;   // 4*16*8*32*16 bf16 = 512 KB

    prepack_weights<<<64, 256, 0, stream>>>(Wq, Wk, Wv, Wo, wfrags);
    lateral_coupling_kernel<<<BT / TOK, NW * 32, SMEM_BYTES, stream>>>(
        h, wfrags, Wlat, gate, (float*)d_out);
}
